// GeometricLayer_19430432047359
// MI455X (gfx1250) — matrix operand order repaired and checked by reading, not compiled, not run on hardware
//
#include <hip/hip_runtime.h>
#include <math.h>

// ---------------- problem constants ----------------
constexpr int BB   = 8;
constexpr int NN   = 8192;
constexpr int MM   = 2048;
constexpr int KK   = 16;
constexpr int DIMC = 64;
constexpr int HIDC = 128;
constexpr int NG   = 8;

typedef _Float16 half_t;
typedef __attribute__((ext_vector_type(16))) _Float16 v16h;
typedef __attribute__((ext_vector_type(8)))  float    v8f;

// ---------------- WMMA helpers (CDNA5 16x16x32 f16 -> f32) ----------------
// A-frag (16x32, MxK): lane row = M = lane&15 ; element e<8 -> k = 8*hi + k0 + e
//                      element e>=8 -> k = 8*hi + k0 + 16 + (e-8)
__device__ __forceinline__ v16h make_afrag_h(const half_t* base, int k0) {
  v16h a;
#pragma unroll
  for (int e = 0; e < 8; ++e) a[e] = base[k0 + e];
#pragma unroll
  for (int e = 0; e < 8; ++e) a[8 + e] = base[k0 + 16 + e];
  return a;
}
__device__ __forceinline__ v16h make_afrag_w(const float* base, int k0) {
  v16h a;
#pragma unroll
  for (int e = 0; e < 8; ++e) a[e] = (half_t)base[k0 + e];
#pragma unroll
  for (int e = 0; e < 8; ++e) a[8 + e] = (half_t)base[k0 + 16 + e];
  return a;
}
// B-frag (32x16, KxN): lane col = lane&15 ; element e<8 -> k = 8*hi + k0' + e
//                      element e>=8 -> k = 8*hi + k0' + 16 + (e-8)
__device__ __forceinline__ v16h make_bfrag(const half_t* base, int k0) {
  v16h b;
#pragma unroll
  for (int e = 0; e < 8; ++e) b[e] = base[k0 + e];
#pragma unroll
  for (int e = 0; e < 8; ++e) b[8 + e] = base[k0 + 16 + e];
  return b;
}
__device__ __forceinline__ v8f wmma_f16(v16h a, v16h b, v8f c) {
  return __builtin_amdgcn_wmma_f32_16x16x32_f16(false, a, false, b, (short)0, c,
                                                false, false);
}

// ---------------- wave32 shuffle reductions ----------------
__device__ __forceinline__ float red_add16(float v) {
#pragma unroll
  for (int o = 8; o >= 1; o >>= 1) v += __shfl_xor(v, o, 32);
  return v;
}
__device__ __forceinline__ float red_max16(float v) {
#pragma unroll
  for (int o = 8; o >= 1; o >>= 1) v = fmaxf(v, __shfl_xor(v, o, 32));
  return v;
}
__device__ __forceinline__ float red_add32(float v) {
#pragma unroll
  for (int o = 16; o >= 1; o >>= 1) v += __shfl_xor(v, o, 32);
  return v;
}

// ---------------- workspace layout ----------------
constexpr size_t alup(size_t x) { return (x + 255) & ~(size_t)255; }
constexpr size_t SZ_FEAT = alup((size_t)BB * MM * DIMC * sizeof(half_t));
constexpr size_t SZ_SQN  = alup((size_t)BB * MM * sizeof(float));
constexpr size_t SZ_FIDX = alup((size_t)BB * MM * KK * sizeof(int));
constexpr size_t SZ_QF   = SZ_FEAT;
constexpr size_t SZ_KVF  = alup((size_t)BB * NN * DIMC * sizeof(half_t));
constexpr size_t SZ_PEO  = alup((size_t)BB * MM * KK * DIMC * sizeof(half_t));
constexpr size_t SZ_FGT  = alup((size_t)BB * MM * DIMC * sizeof(float));
constexpr size_t SZ_SH   = alup((size_t)BB * MM * HIDC * sizeof(half_t));
constexpr size_t SZ_BUF  = alup((size_t)BB * MM * KK * HIDC * sizeof(half_t));
constexpr size_t SZ_PART = alup((size_t)BB * MM * NG * 2 * sizeof(float));
constexpr size_t SZ_STAT = alup((size_t)BB * NG * 2 * sizeof(float));

constexpr size_t OFF_FEAT = 0;
constexpr size_t OFF_SQN  = OFF_FEAT + SZ_FEAT;
constexpr size_t OFF_FIDX = OFF_SQN + SZ_SQN;
constexpr size_t OFF_QF   = OFF_FIDX + SZ_FIDX;
constexpr size_t OFF_KVF  = OFF_QF + SZ_QF;
constexpr size_t OFF_PEO  = OFF_KVF + SZ_KVF;
constexpr size_t OFF_FGT  = OFF_PEO + SZ_PEO;
constexpr size_t OFF_SH   = OFF_FGT + SZ_FGT;
constexpr size_t OFF_BUFA = OFF_SH + SZ_SH;    // h1, later reused for attn hidden
constexpr size_t OFF_BUFB = OFF_BUFA + SZ_BUF; // pe hidden
constexpr size_t OFF_P1   = OFF_BUFB + SZ_BUF; // pre-norm partials
constexpr size_t OFF_P2   = OFF_P1 + SZ_PART;  // local
constexpr size_t OFF_P3   = OFF_P2 + SZ_PART;  // attn
constexpr size_t OFF_P4   = OFF_P3 + SZ_PART;  // sem
constexpr size_t OFF_S1   = OFF_P4 + SZ_PART;
constexpr size_t OFF_S2   = OFF_S1 + SZ_STAT;
constexpr size_t OFF_S3   = OFF_S2 + SZ_STAT;
constexpr size_t OFF_S4   = OFF_S3 + SZ_STAT;

// =================================================================
// 1) pos_emb[b,:,m] = W(4->64) @ sum_k pattern + K*bias ; also f16 feat + |.|^2
// =================================================================
__global__ void __launch_bounds__(64) k_posemb(const float* __restrict__ qx,
                                               const float* __restrict__ kx,
                                               const int* __restrict__ knn,
                                               const float* __restrict__ Wp,
                                               const float* __restrict__ bp,
                                               half_t* __restrict__ feat,
                                               float* __restrict__ sqn) {
  int bm = blockIdx.x;
  int b = bm / MM, m = bm % MM;
  int t = threadIdx.x;
  __shared__ float patk[KK][4];
  __shared__ float pat[4];
  __shared__ float red2[2];
  if (t < KK) {
    int idx = knn[(size_t)bm * KK + t];
    float dx = kx[(b * 3 + 0) * NN + idx] - qx[(b * 3 + 0) * MM + m];
    float dy = kx[(b * 3 + 1) * NN + idx] - qx[(b * 3 + 1) * MM + m];
    float dz = kx[(b * 3 + 2) * NN + idx] - qx[(b * 3 + 2) * MM + m];
    float dist = sqrtf(dx * dx + dy * dy + dz * dz);
    float inv = 1.0f / fmaxf(dist, 1e-12f);
    patk[t][0] = dx * inv; patk[t][1] = dy * inv;
    patk[t][2] = dz * inv; patk[t][3] = dist;
  }
  __syncthreads();
  if (t < 4) {
    float s = 0.f;
    for (int k = 0; k < KK; ++k) s += patk[k][t];
    pat[t] = s;
  }
  __syncthreads();
  float v = Wp[t * 4 + 0] * pat[0] + Wp[t * 4 + 1] * pat[1] +
            Wp[t * 4 + 2] * pat[2] + Wp[t * 4 + 3] * pat[3] + (float)KK * bp[t];
  feat[(size_t)bm * DIMC + t] = (half_t)v;
  float sq = red_add32(v * v);
  if ((t & 31) == 0) red2[t >> 5] = sq;
  __syncthreads();
  if (t == 0) sqn[bm] = red2[0] + red2[1];
}

// =================================================================
// 2) feature-space kNN: WMMA Gram tiles (2x 16x16x32) + top-16 via wave argmax
// =================================================================
__global__ void __launch_bounds__(256) k_gram_topk(const half_t* __restrict__ feat,
                                                   const float* __restrict__ sqn,
                                                   int* __restrict__ fidx) {
  __shared__ half_t pd[16 * MM]; // 64 KB distance stripe (f16 is fine for ranking)
  int blk = blockIdx.x;
  int b = blk / (MM / 16), mt = blk % (MM / 16);
  int mbase = mt * 16;
  int w = threadIdx.x >> 5, lane = threadIdx.x & 31;
  int col = lane & 15, hi = (lane >> 4) & 1;
  const half_t* fb = feat + (size_t)b * MM * DIMC;

  const half_t* arow = fb + (size_t)(mbase + col) * DIMC;
  v16h a0 = make_afrag_h(arow, 8 * hi);
  v16h a1 = make_afrag_h(arow, 8 * hi + 32);
  float sqq[8];
#pragma unroll
  for (int r = 0; r < 8; ++r) sqq[r] = sqn[b * MM + mbase + r + 8 * hi];

  for (int tile = w; tile < MM / 16; tile += 8) {
    int nb = tile * 16;
    const half_t* brow = fb + (size_t)(nb + col) * DIMC;
    if (tile + 8 < MM / 16)
      __builtin_prefetch(fb + (size_t)((tile + 8) * 16 + col) * DIMC, 0, 0);
    v16h b0 = make_bfrag(brow, 8 * hi);
    v16h b1 = make_bfrag(brow, 8 * hi + 32);
    v8f acc = {};
    acc = wmma_f16(a0, b0, acc);
    acc = wmma_f16(a1, b1, acc);
    float sk = sqn[b * MM + nb + col];
#pragma unroll
    for (int r = 0; r < 8; ++r) {
      float pdv = 2.0f * acc[r] - sqq[r] - sk; // -||xi-xj||^2
      pd[(r + 8 * hi) * MM + nb + col] = (half_t)pdv;
    }
  }
  __syncthreads();

  // selection: wave w owns rows 2w, 2w+1 ; 16 argmax passes each
  for (int s = 0; s < 2; ++s) {
    int row = 2 * w + s;
    for (int it = 0; it < KK; ++it) {
      float bv = -3.0e38f; int bj = 0;
      for (int j = lane; j < MM; j += 32) {
        float v = (float)pd[row * MM + j];
        if (v > bv || (v == bv && j < bj)) { bv = v; bj = j; }
      }
#pragma unroll
      for (int o = 16; o >= 1; o >>= 1) {
        float ov = __shfl_xor(bv, o, 32);
        int   oj = __shfl_xor(bj, o, 32);
        if (ov > bv || (ov == bv && oj < bj)) { bv = ov; bj = oj; }
      }
      if (lane == 0) {
        fidx[(size_t)(b * MM + mbase + row) * KK + it] = bj;
        pd[row * MM + bj] = (half_t)(-3.0e4f);
      }
    }
  }
}

// =================================================================
// 3) qkv conv: out[b,col,c](f16) = qkv_w(64x64) @ x[b,:,col] + b  (L = M or N)
// =================================================================
__global__ void __launch_bounds__(128) k_gemm64(const float* __restrict__ x,
                                                const float* __restrict__ W,
                                                const float* __restrict__ bias,
                                                half_t* __restrict__ out, int L) {
  __shared__ half_t Xt[16][72];
  int blk = blockIdx.x;
  int b = blk / (L / 16), ct = blk % (L / 16);
  int cbase = ct * 16;
  int t = threadIdx.x;
  for (int i = t; i < 16 * DIMC; i += 128) {
    int c = i >> 6, ci = i & 63;
    Xt[c][ci] = (half_t)x[((size_t)b * DIMC + ci) * L + cbase + c];
  }
  __syncthreads();
  int w = t >> 5, lane = t & 31, col = lane & 15, hi = (lane >> 4) & 1;
  v8f acc = {};
#pragma unroll
  for (int ch = 0; ch < 2; ++ch) {
    v16h a = make_afrag_w(W + (size_t)(w * 16 + col) * DIMC, 8 * hi + 32 * ch);
    v16h bf = make_bfrag(&Xt[col][0], 8 * hi + 32 * ch);
    acc = wmma_f16(a, bf, acc);
  }
#pragma unroll
  for (int r = 0; r < 8; ++r) {
    int row = w * 16 + r + 8 * hi;
    out[((size_t)b * L + cbase + col) * DIMC + row] = (half_t)(acc[r] + bias[row]);
  }
}

// =================================================================
// 4) pe hidden: W(3->128) @ (q - knn_xyz) + b ; per-group partial stats
// =================================================================
__global__ void __launch_bounds__(128) k_peh(const float* __restrict__ qx,
                                             const float* __restrict__ kx,
                                             const int* __restrict__ knn,
                                             const float* __restrict__ W1,
                                             const float* __restrict__ b1,
                                             half_t* __restrict__ peh,
                                             float* __restrict__ part) {
  int bm = blockIdx.x;
  int b = bm / MM, m = bm % MM;
  int c = threadIdx.x;
  float w0 = W1[c * 3 + 0], w1 = W1[c * 3 + 1], w2 = W1[c * 3 + 2], bc = b1[c];
  float q0 = qx[(b * 3 + 0) * MM + m];
  float q1 = qx[(b * 3 + 1) * MM + m];
  float q2 = qx[(b * 3 + 2) * MM + m];
  float s = 0.f, ss = 0.f;
  for (int k = 0; k < KK; ++k) {
    int idx = knn[(size_t)bm * KK + k];
    float d0 = q0 - kx[(b * 3 + 0) * NN + idx];
    float d1 = q1 - kx[(b * 3 + 1) * NN + idx];
    float d2 = q2 - kx[(b * 3 + 2) * NN + idx];
    float v = w0 * d0 + w1 * d1 + w2 * d2 + bc;
    peh[((size_t)bm * KK + k) * HIDC + c] = (half_t)v;
    s += v; ss += v * v;
  }
  s = red_add16(s); ss = red_add16(ss);
  if ((c & 15) == 0) {
    int g = c >> 4;
    part[((size_t)bm * NG + g) * 2 + 0] = s;
    part[((size_t)bm * NG + g) * 2 + 1] = ss;
  }
}

// =================================================================
// 5) graph-feature MLP layer 1: h1 = local_w1(129->128) @ [ctr;nb;nrm] + b1
// =================================================================
__global__ void __launch_bounds__(256) k_intra1(const half_t* __restrict__ feat,
                                                const int* __restrict__ fidx,
                                                const float* __restrict__ W1,
                                                const float* __restrict__ b1,
                                                half_t* __restrict__ h1,
                                                float* __restrict__ part) {
  __shared__ half_t Xt[16][136]; // cin 0..63 ctr, 64..127 nb, 128 nrm
  int bm = blockIdx.x;
  int b = bm / MM;
  int t = threadIdx.x;
  {
    int k = t >> 4, c0 = t & 15; // half-wave per edge
    int nidx = fidx[(size_t)bm * KK + k];
    const half_t* ctr = feat + (size_t)bm * DIMC;
    const half_t* nbp = feat + ((size_t)b * MM + nidx) * DIMC;
    float sq = 0.f;
    for (int c = c0; c < DIMC; c += 16) {
      float cv = (float)ctr[c], nv = (float)nbp[c];
      Xt[k][c] = (half_t)cv;
      Xt[k][64 + c] = (half_t)nv;
      float d = nv - cv;
      sq += d * d;
    }
    sq = red_add16(sq);
    if (c0 == 0) Xt[k][128] = (half_t)sqrtf(sq);
  }
  __syncthreads();
  int w = t >> 5, lane = t & 31, col = lane & 15, hi = (lane >> 4) & 1;
  int row0 = w * 16; // one norm group per wave (HID/NG == 16)
  v8f acc = {};
#pragma unroll
  for (int ch = 0; ch < 4; ++ch) {
    v16h a = make_afrag_w(W1 + (size_t)(row0 + col) * 129, 8 * hi + 32 * ch);
    v16h bf = make_bfrag(&Xt[col][0], 8 * hi + 32 * ch);
    acc = wmma_f16(a, bf, acc);
  }
  float nrm = (float)Xt[col][128];
  float s = 0.f, ss = 0.f;
#pragma unroll
  for (int r = 0; r < 8; ++r) {
    int row = row0 + r + 8 * hi;
    float v = acc[r] + W1[(size_t)row * 129 + 128] * nrm + b1[row];
    h1[((size_t)bm * KK + col) * HIDC + row] = (half_t)v;
    s += v; ss += v * v;
  }
  s = red_add32(s); ss = red_add32(ss);
  if (lane == 0) {
    part[((size_t)bm * NG + w) * 2 + 0] = s;
    part[((size_t)bm * NG + w) * 2 + 1] = ss;
  }
}

// =================================================================
// 6) finalize group-norm stats: mean, rstd per (b,g)
// =================================================================
__global__ void __launch_bounds__(256) k_finstat(const float* __restrict__ part,
                                                 int npart, float inv_count,
                                                 float* __restrict__ stats) {
  int bg = blockIdx.x;
  int b = bg / NG, g = bg % NG;
  __shared__ float r1[8], r2[8];
  float s = 0.f, ss = 0.f;
  for (int i = threadIdx.x; i < npart; i += 256) {
    const float* p = part + (((size_t)b * npart + i) * NG + g) * 2;
    s += p[0]; ss += p[1];
  }
  s = red_add32(s); ss = red_add32(ss);
  int w = threadIdx.x >> 5, lane = threadIdx.x & 31;
  if (lane == 0) { r1[w] = s; r2[w] = ss; }
  __syncthreads();
  if (threadIdx.x == 0) {
    float S = 0.f, SS = 0.f;
    for (int i = 0; i < 8; ++i) { S += r1[i]; SS += r2[i]; }
    float mean = S * inv_count;
    float var = SS * inv_count - mean * mean;
    stats[bg * 2 + 0] = mean;
    stats[bg * 2 + 1] = rsqrtf(fmaxf(var, 0.f) + 1e-5f);
  }
}

// =================================================================
// 7) pe layer 2: pe = pre_w2(128->64) @ relu(gn(peh)) + b2  -> f16 [b,m,k,64]
// =================================================================
__global__ void __launch_bounds__(128) k_pe2(const half_t* __restrict__ peh,
                                             const float* __restrict__ stats,
                                             const float* __restrict__ gamma,
                                             const float* __restrict__ beta,
                                             const float* __restrict__ W2,
                                             const float* __restrict__ b2,
                                             half_t* __restrict__ peo) {
  __shared__ half_t Xt[16][136];
  int bm = blockIdx.x;
  int b = bm / MM;
  int t = threadIdx.x;
  for (int i = t; i < 16 * HIDC; i += 128) {
    int k = i >> 7, c = i & 127, g = c >> 4;
    float mean = stats[(b * NG + g) * 2], rstd = stats[(b * NG + g) * 2 + 1];
    float v = ((float)peh[((size_t)bm * KK + k) * HIDC + c] - mean) * rstd *
                  gamma[c] + beta[c];
    Xt[k][c] = (half_t)fmaxf(v, 0.f);
  }
  __syncthreads();
  int w = t >> 5, lane = t & 31, col = lane & 15, hi = (lane >> 4) & 1;
  v8f acc = {};
#pragma unroll
  for (int ch = 0; ch < 4; ++ch) {
    v16h a = make_afrag_w(W2 + (size_t)(w * 16 + col) * HIDC, 8 * hi + 32 * ch);
    v16h bf = make_bfrag(&Xt[col][0], 8 * hi + 32 * ch);
    acc = wmma_f16(a, bf, acc);
  }
#pragma unroll
  for (int r = 0; r < 8; ++r) {
    int row = w * 16 + r + 8 * hi;
    peo[((size_t)bm * KK + col) * DIMC + row] = (half_t)(acc[r] + b2[row]);
  }
}

// =================================================================
// 8) intra layer 2: relu(gn(h1)) -> local_w2(128->64) -> max over K
// =================================================================
__global__ void __launch_bounds__(128) k_intra2(const half_t* __restrict__ h1,
                                                const float* __restrict__ stats,
                                                const float* __restrict__ gamma,
                                                const float* __restrict__ beta,
                                                const float* __restrict__ W2,
                                                const float* __restrict__ b2,
                                                float* __restrict__ out) {
  __shared__ half_t Xt[16][136];
  int bm = blockIdx.x;
  int b = bm / MM, m = bm % MM;
  int t = threadIdx.x;
  for (int i = t; i < 16 * HIDC; i += 128) {
    int k = i >> 7, c = i & 127, g = c >> 4;
    float mean = stats[(b * NG + g) * 2], rstd = stats[(b * NG + g) * 2 + 1];
    float v = ((float)h1[((size_t)bm * KK + k) * HIDC + c] - mean) * rstd *
                  gamma[c] + beta[c];
    Xt[k][c] = (half_t)fmaxf(v, 0.f);
  }
  __syncthreads();
  int w = t >> 5, lane = t & 31, col = lane & 15, hi = (lane >> 4) & 1;
  v8f acc = {};
#pragma unroll
  for (int ch = 0; ch < 4; ++ch) {
    v16h a = make_afrag_w(W2 + (size_t)(w * 16 + col) * HIDC, 8 * hi + 32 * ch);
    v16h bf = make_bfrag(&Xt[col][0], 8 * hi + 32 * ch);
    acc = wmma_f16(a, bf, acc);
  }
#pragma unroll
  for (int r = 0; r < 8; ++r) {
    int row = w * 16 + r + 8 * hi;
    float v = red_max16(acc[r] + b2[row]); // max over the 16 edges
    if (col == 0) out[((size_t)b * DIMC + row) * MM + m] = v;
  }
}

// =================================================================
// 9) attn layer 1: attn_w1(64->128) @ (query - keyf + pe) + b1
// =================================================================
__global__ void __launch_bounds__(256) k_attn1(const half_t* __restrict__ qf,
                                               const half_t* __restrict__ kvf,
                                               const int* __restrict__ knn,
                                               const half_t* __restrict__ peo,
                                               const float* __restrict__ W1,
                                               const float* __restrict__ b1,
                                               half_t* __restrict__ ah,
                                               float* __restrict__ part) {
  __shared__ half_t Xt[16][72];
  __shared__ int sidx[16];
  int bm = blockIdx.x;
  int b = bm / MM;
  int t = threadIdx.x;
  if (t < 16) sidx[t] = knn[(size_t)bm * KK + t];
  __syncthreads();
  for (int i = t; i < 16 * DIMC; i += 256) {
    int k = i >> 6, c = i & 63;
    float q = (float)qf[(size_t)bm * DIMC + c];
    float kf = (float)kvf[((size_t)b * NN + sidx[k]) * DIMC + c];
    float pe = (float)peo[((size_t)bm * KK + k) * DIMC + c];
    Xt[k][c] = (half_t)(q - kf + pe);
  }
  __syncthreads();
  int w = t >> 5, lane = t & 31, col = lane & 15, hi = (lane >> 4) & 1;
  v8f acc = {};
#pragma unroll
  for (int ch = 0; ch < 2; ++ch) {
    v16h a = make_afrag_w(W1 + (size_t)(w * 16 + col) * DIMC, 8 * hi + 32 * ch);
    v16h bf = make_bfrag(&Xt[col][0], 8 * hi + 32 * ch);
    acc = wmma_f16(a, bf, acc);
  }
  float s = 0.f, ss = 0.f;
#pragma unroll
  for (int r = 0; r < 8; ++r) {
    int row = w * 16 + r + 8 * hi;
    float v = acc[r] + b1[row];
    ah[((size_t)bm * KK + col) * HIDC + row] = (half_t)v;
    s += v; ss += v * v;
  }
  s = red_add32(s); ss = red_add32(ss);
  if (lane == 0) {
    part[((size_t)bm * NG + w) * 2 + 0] = s;
    part[((size_t)bm * NG + w) * 2 + 1] = ss;
  }
}

// =================================================================
// 10) attn layer 2 + mask + softmax(K) + fgt = sum_k a*(value+pe)
// =================================================================
__global__ void __launch_bounds__(128) k_attn2(const half_t* __restrict__ ah,
                                               const float* __restrict__ stats,
                                               const float* __restrict__ gamma,
                                               const float* __restrict__ beta,
                                               const float* __restrict__ W2,
                                               const float* __restrict__ b2,
                                               const unsigned char* __restrict__ mask,
                                               const int* __restrict__ knn,
                                               const half_t* __restrict__ kvf,
                                               const half_t* __restrict__ peo,
                                               float* __restrict__ fgt) {
  __shared__ half_t Xt[16][136];
  int bm = blockIdx.x;
  int b = bm / MM;
  int t = threadIdx.x;
  for (int i = t; i < 16 * HIDC; i += 128) {
    int k = i >> 7, c = i & 127, g = c >> 4;
    float mean = stats[(b * NG + g) * 2], rstd = stats[(b * NG + g) * 2 + 1];
    float v = ((float)ah[((size_t)bm * KK + k) * HIDC + c] - mean) * rstd *
                  gamma[c] + beta[c];
    Xt[k][c] = (half_t)fmaxf(v, 0.f);
  }
  __syncthreads();
  int w = t >> 5, lane = t & 31, col = lane & 15, hi = (lane >> 4) & 1;
  v8f acc = {};
#pragma unroll
  for (int ch = 0; ch < 4; ++ch) {
    v16h a = make_afrag_w(W2 + (size_t)(w * 16 + col) * HIDC, 8 * hi + 32 * ch);
    v16h bf = make_bfrag(&Xt[col][0], 8 * hi + 32 * ch);
    acc = wmma_f16(a, bf, acc);
  }
  const float scale = 0.125f; // 1/sqrt(DIM)
  bool mk = mask[(size_t)bm * KK + col] != 0;
  int nidx = knn[(size_t)bm * KK + col];
  float av[8];
#pragma unroll
  for (int r = 0; r < 8; ++r) {
    int row = w * 16 + r + 8 * hi;
    float v = (acc[r] + b2[row]) * scale;
    av[r] = mk ? v : -3.402823e38f;
  }
#pragma unroll
  for (int r = 0; r < 8; ++r) { // softmax over the 16 edge columns
    float mx = red_max16(av[r]);
    float e = __expf(av[r] - mx);
    float sum = red_add16(e);
    av[r] = e / sum;
  }
#pragma unroll
  for (int r = 0; r < 8; ++r) {
    int row = w * 16 + r + 8 * hi;
    float val = (float)kvf[((size_t)b * NN + nidx) * DIMC + row] +
                (float)peo[((size_t)bm * KK + col) * DIMC + row];
    float contrib = red_add16(av[r] * val);
    if (col == 0) fgt[(size_t)bm * DIMC + row] = contrib;
  }
}

// =================================================================
// 11) sem layer 1: sem_w1(64->128) @ fgt + b1 ; partial stats
// =================================================================
__global__ void __launch_bounds__(256) k_sem1(const float* __restrict__ fgt,
                                              const float* __restrict__ W1,
                                              const float* __restrict__ b1,
                                              half_t* __restrict__ sh,
                                              float* __restrict__ part) {
  __shared__ half_t Xt[16][72];
  int blk = blockIdx.x;
  int b = blk / (MM / 16), mt = blk % (MM / 16);
  int mb = mt * 16;
  int t = threadIdx.x;
  for (int i = t; i < 16 * DIMC; i += 256) {
    int cc = i >> 6, c = i & 63;
    Xt[cc][c] = (half_t)fgt[((size_t)b * MM + mb + cc) * DIMC + c];
  }
  __syncthreads();
  int w = t >> 5, lane = t & 31, col = lane & 15, hi = (lane >> 4) & 1;
  v8f acc = {};
#pragma unroll
  for (int ch = 0; ch < 2; ++ch) {
    v16h a = make_afrag_w(W1 + (size_t)(w * 16 + col) * DIMC, 8 * hi + 32 * ch);
    v16h bf = make_bfrag(&Xt[col][0], 8 * hi + 32 * ch);
    acc = wmma_f16(a, bf, acc);
  }
  float s = 0.f, ss = 0.f;
#pragma unroll
  for (int r = 0; r < 8; ++r) {
    int row = w * 16 + r + 8 * hi;
    float v = acc[r] + b1[row];
    sh[((size_t)b * MM + mb + col) * HIDC + row] = (half_t)v;
    s += v; ss += v * v;
  }
  s = red_add32(s); ss = red_add32(ss);
  if (lane == 0) {
    part[((size_t)blk * NG + w) * 2 + 0] = s;
    part[((size_t)blk * NG + w) * 2 + 1] = ss;
  }
}

// =================================================================
// 12) sem layer 2: inter = sem_w2(128->64) @ relu(gn(sh)) + b2
// =================================================================
__global__ void __launch_bounds__(128) k_sem2(const half_t* __restrict__ sh,
                                              const float* __restrict__ stats,
                                              const float* __restrict__ gamma,
                                              const float* __restrict__ beta,
                                              const float* __restrict__ W2,
                                              const float* __restrict__ b2,
                                              float* __restrict__ out) {
  __shared__ half_t Xt[16][136];
  int blk = blockIdx.x;
  int b = blk / (MM / 16), mt = blk % (MM / 16);
  int mb = mt * 16;
  int t = threadIdx.x;
  for (int i = t; i < 16 * HIDC; i += 128) {
    int cc = i >> 7, c = i & 127, g = c >> 4;
    float mean = stats[(b * NG + g) * 2], rstd = stats[(b * NG + g) * 2 + 1];
    float v = ((float)sh[((size_t)b * MM + mb + cc) * HIDC + c] - mean) * rstd *
                  gamma[c] + beta[c];
    Xt[cc][c] = (half_t)fmaxf(v, 0.f);
  }
  __syncthreads();
  int w = t >> 5, lane = t & 31, col = lane & 15, hi = (lane >> 4) & 1;
  v8f acc = {};
#pragma unroll
  for (int ch = 0; ch < 4; ++ch) {
    v16h a = make_afrag_w(W2 + (size_t)(w * 16 + col) * HIDC, 8 * hi + 32 * ch);
    v16h bf = make_bfrag(&Xt[col][0], 8 * hi + 32 * ch);
    acc = wmma_f16(a, bf, acc);
  }
#pragma unroll
  for (int r = 0; r < 8; ++r) {
    int row = w * 16 + r + 8 * hi;
    out[((size_t)b * DIMC + row) * MM + mb + col] = acc[r] + b2[row];
  }
}

// =================================================================
extern "C" void kernel_launch(void* const* d_in, const int* in_sizes, int n_in,
                              void* d_out, int out_size, void* d_ws, size_t ws_size,
                              hipStream_t stream) {
  (void)in_sizes; (void)n_in; (void)out_size; (void)ws_size;
  const float* q_xyzs    = (const float*)d_in[0];
  const float* k_xyzs    = (const float*)d_in[1];
  const float* sam_feats = (const float*)d_in[2];
  const float* xyz_feats = (const float*)d_in[3];
  const int*   knn_idx   = (const int*)d_in[4];
  const unsigned char* mask = (const unsigned char*)d_in[5];
  const float* pre_nn_w = (const float*)d_in[6];  const float* pre_nn_b = (const float*)d_in[7];
  const float* local_w1 = (const float*)d_in[8];  const float* local_b1 = (const float*)d_in[9];
  const float* local_g1 = (const float*)d_in[10]; const float* local_be1= (const float*)d_in[11];
  const float* local_w2 = (const float*)d_in[12]; const float* local_b2 = (const float*)d_in[13];
  const float* qkv_w    = (const float*)d_in[14]; const float* qkv_b    = (const float*)d_in[15];
  const float* pre_w1   = (const float*)d_in[16]; const float* pre_b1   = (const float*)d_in[17];
  const float* pre_g    = (const float*)d_in[18]; const float* pre_be   = (const float*)d_in[19];
  const float* pre_w2   = (const float*)d_in[20]; const float* pre_b2   = (const float*)d_in[21];
  const float* attn_w1  = (const float*)d_in[22]; const float* attn_b1  = (const float*)d_in[23];
  const float* attn_g   = (const float*)d_in[24]; const float* attn_be  = (const float*)d_in[25];
  const float* attn_w2  = (const float*)d_in[26]; const float* attn_b2  = (const float*)d_in[27];
  const float* sem_w1   = (const float*)d_in[28]; const float* sem_b1   = (const float*)d_in[29];
  const float* sem_g    = (const float*)d_in[30]; const float* sem_be   = (const float*)d_in[31];
  const float* sem_w2   = (const float*)d_in[32]; const float* sem_b2   = (const float*)d_in[33];

  char* ws = (char*)d_ws;
  half_t* feat = (half_t*)(ws + OFF_FEAT);
  float*  sqn  = (float*)(ws + OFF_SQN);
  int*    fidx = (int*)(ws + OFF_FIDX);
  half_t* qf   = (half_t*)(ws + OFF_QF);
  half_t* kvf  = (half_t*)(ws + OFF_KVF);
  half_t* peo  = (half_t*)(ws + OFF_PEO);
  float*  fgt  = (float*)(ws + OFF_FGT);
  half_t* sh   = (half_t*)(ws + OFF_SH);
  half_t* h1   = (half_t*)(ws + OFF_BUFA); // reused for attn hidden afterwards
  half_t* ah   = (half_t*)(ws + OFF_BUFA);
  half_t* peh  = (half_t*)(ws + OFF_BUFB);
  float*  p_pe = (float*)(ws + OFF_P1);
  float*  p_lc = (float*)(ws + OFF_P2);
  float*  p_at = (float*)(ws + OFF_P3);
  float*  p_sm = (float*)(ws + OFF_P4);
  float*  s_pe = (float*)(ws + OFF_S1);
  float*  s_lc = (float*)(ws + OFF_S2);
  float*  s_at = (float*)(ws + OFF_S3);
  float*  s_sm = (float*)(ws + OFF_S4);

  float* intra = (float*)d_out;
  float* inter = (float*)d_out + (size_t)BB * DIMC * MM;

  const float invMK = 1.0f / (16.0f * (float)MM * (float)KK);
  const float invM  = 1.0f / (16.0f * (float)MM);

  k_posemb<<<BB * MM, 64, 0, stream>>>(q_xyzs, k_xyzs, knn_idx, pre_nn_w, pre_nn_b,
                                       feat, sqn);
  k_gram_topk<<<BB * (MM / 16), 256, 0, stream>>>(feat, sqn, fidx);
  k_gemm64<<<BB * (MM / 16), 128, 0, stream>>>(sam_feats, qkv_w, qkv_b, qf, MM);
  k_gemm64<<<BB * (NN / 16), 128, 0, stream>>>(xyz_feats, qkv_w, qkv_b, kvf, NN);
  k_peh<<<BB * MM, 128, 0, stream>>>(q_xyzs, k_xyzs, knn_idx, pre_w1, pre_b1, peh, p_pe);
  k_finstat<<<BB * NG, 256, 0, stream>>>(p_pe, MM, invMK, s_pe);
  k_pe2<<<BB * MM, 128, 0, stream>>>(peh, s_pe, pre_g, pre_be, pre_w2, pre_b2, peo);
  k_intra1<<<BB * MM, 256, 0, stream>>>(feat, fidx, local_w1, local_b1, h1, p_lc);
  k_finstat<<<BB * NG, 256, 0, stream>>>(p_lc, MM, invMK, s_lc);
  k_intra2<<<BB * MM, 128, 0, stream>>>(h1, s_lc, local_g1, local_be1, local_w2,
                                        local_b2, intra);
  k_attn1<<<BB * MM, 256, 0, stream>>>(qf, kvf, knn_idx, peo, attn_w1, attn_b1, ah, p_at);
  k_finstat<<<BB * NG, 256, 0, stream>>>(p_at, MM, invMK, s_at);
  k_attn2<<<BB * MM, 128, 0, stream>>>(ah, s_at, attn_g, attn_be, attn_w2, attn_b2,
                                       mask, knn_idx, kvf, peo, fgt);
  k_sem1<<<BB * (MM / 16), 256, 0, stream>>>(fgt, sem_w1, sem_b1, sh, p_sm);
  k_finstat<<<BB * NG, 256, 0, stream>>>(p_sm, MM / 16, invM, s_sm);
  k_sem2<<<BB * (MM / 16), 128, 0, stream>>>(sh, s_sm, sem_g, sem_be, sem_w2, sem_b2,
                                             inter);
}
